// GaussianDecoder_40982577938710
// MI455X (gfx1250) — matrix operand order repaired and checked by reading, not compiled, not run on hardware
//
#include <hip/hip_runtime.h>

// ---------------------------------------------------------------------------
// Types for CDNA5 WMMA / TDM
// ---------------------------------------------------------------------------
typedef __attribute__((ext_vector_type(16))) __bf16          v16bf;
typedef __attribute__((ext_vector_type(2)))  __bf16          v2bf;
typedef __attribute__((ext_vector_type(8)))  float           v8f;
typedef __attribute__((ext_vector_type(16))) unsigned short  v16u;
typedef __attribute__((ext_vector_type(8)))  unsigned short  v8u;
typedef __attribute__((ext_vector_type(8)))  unsigned int    v8ui;
typedef __attribute__((ext_vector_type(4)))  unsigned int    u32x4;
typedef __attribute__((ext_vector_type(8)))  int             i32x8;
typedef __attribute__((ext_vector_type(4)))  int             i32x4;

#define HID        256
#define KCHUNKS    8          // 256 / 32
#define TILE_ROWS  256        // rows per workgroup
#define THREADS    512        // 16 wave32
#define NPTS       8192       // points per batch (mask is all-false)
#define RANK       32
#define ACC_STRIDE 1064       // 1024 (C) + 32 (t) + 2 (sums) + pad

// ws layout (ushort units for weights)
#define W_MLP_SZ   (65536 + 65536 + 4096)   // mean / d region (W3 padded to 16 cols)
#define W_V_SZ     (65536 + 65536 + 8192)   // V region (W3 = 32 cols)
#define WS_MEAN    0
#define WS_D       (W_MLP_SZ)
#define WS_V       (2 * W_MLP_SZ)
#define WS_ACC_U   (2 * W_MLP_SZ + W_V_SZ)  // ushort offset where accum (float) starts

// hardware bf16 convert (v_cvt_*bf16_f32 class) instead of manual bit-twiddle
static __device__ __forceinline__ unsigned short f2bf(float f) {
  return __builtin_bit_cast(unsigned short, (__bf16)f);
}
static __device__ __forceinline__ unsigned int f2bf2(float lo, float hi) {
  v2bf p;
  p[0] = (__bf16)lo;
  p[1] = (__bf16)hi;
  return __builtin_bit_cast(unsigned int, p);
}

static __device__ __forceinline__ v8f wmma_bf16(v16u a, v16u b, v8f c) {
  return __builtin_amdgcn_wmma_f32_16x16x32_bf16(
      false, __builtin_bit_cast(v16bf, a),
      false, __builtin_bit_cast(v16bf, b),
      (short)0, c, false, false);
}

// ---------------------------------------------------------------------------
// Tensor Data Mover: 1-D tile (n8 x 8B elements) global -> LDS.
// D# per CDNA5 ISA:
//   group0: count=1 | lds_addr | global_addr[56:0] | type=2
//   group1: data_size=3(8B), tensor_dim0=tile_dim0=n8, tensor_dim1=1,
//           tensor_dim0_stride=n8
// This toolchain's builtin takes 6 args (g0, g1, g2, g3, g4, cpol).
// Issued by one wave only; completion via s_wait_tensorcnt 0.
// ---------------------------------------------------------------------------
static __device__ __forceinline__ void tdm_load(const void* gsrc,
                                                unsigned lds_off,
                                                unsigned nbytes) {
  unsigned long long ga = (unsigned long long)(size_t)gsrc;
  unsigned n8 = nbytes >> 3;                               // 8-byte elements
  u32x4 g0;
  g0[0] = 1u;                                              // count=1 (valid), user mode
  g0[1] = lds_off;                                         // lds_addr (bytes)
  g0[2] = (unsigned)(ga & 0xFFFFFFFFu);                    // global_addr[31:0]
  g0[3] = (unsigned)((ga >> 32) & 0x01FFFFFFu) | 0x80000000u; // addr[56:32] | type=2
  i32x8 g1;
  g1[0] = 3 << 16;                                         // data_size = 8B
  g1[1] = (int)(n8 << 16);                                 // tensor_dim0[15:0]
  g1[2] = (int)((n8 >> 16) & 0xFFFFu) | (1 << 16);         // tensor_dim0[31:16] | tensor_dim1=1
  g1[3] = (int)(n8 << 16);                                 // tensor_dim1 hi=0 | tile_dim0=n8
  g1[4] = 0;                                               // tile_dim1=0, tile_dim2=0
  g1[5] = (int)n8;                                         // tensor_dim0_stride[31:0]
  g1[6] = 0;
  g1[7] = 0;
  i32x4 z4 = {0, 0, 0, 0};
  i32x8 z8 = {0, 0, 0, 0, 0, 0, 0, 0};
  __builtin_amdgcn_tensor_load_to_lds(g0, g1, z4, z4, z8, 0);
}

// B fragment: 32 contiguous bytes per lane (pre-swizzled by gd_prep)
static __device__ __forceinline__ v16u ldsBfrag(const unsigned short* wb,
                                                int nt, int kc, int lane) {
  return *(const v16u*)(wb + ((((nt * KCHUNKS) + kc) * 32 + lane) << 4));
}

// A fragment from this wave's row-major bf16 activations in LDS
static __device__ __forceinline__ v16u ldsAfrag(const unsigned short* actsw,
                                                int m, int g, int kc) {
  const v8u* p = (const v8u*)(actsw + m * HID + kc * 32 + 8 * g);
  v8u lo = p[0];        // K = kc*32 + 8g .. +7
  v8u hi = p[2];        // K = kc*32 + 16 + 8g .. +7
  return __builtin_shufflevector(lo, hi, 0, 1, 2, 3, 4, 5, 6, 7,
                                 8, 9, 10, 11, 12, 13, 14, 15);
}

// stage weights into LDS via the Tensor Data Mover (wave 0 issues + waits)
static __device__ __forceinline__ void stageW_tdm(const unsigned short* src,
                                                  unsigned nbytes, int tid) {
  if (tid == 0) tdm_load(src, 0u, nbytes);
  if (tid < 32) __builtin_amdgcn_s_wait_tensorcnt(0);
  __syncthreads();
}

// X @ W + b, relu, result (bf16) into this wave's LDS activation slot
static __device__ __forceinline__ void hiddenLayer(const v16u* frag,
                                                   const unsigned short* wbuf,
                                                   const float* __restrict__ bias,
                                                   unsigned short* actsw, int lane) {
  const int n  = lane & 15;
  const int hi = lane >> 4;
  for (int nt = 0; nt < 16; ++nt) {
    v8f c = {};
#pragma unroll
    for (int kc = 0; kc < KCHUNKS; ++kc)
      c = wmma_bf16(frag[kc], ldsBfrag(wbuf, nt, kc, lane), c);
    float b = bias[nt * 16 + n];
#pragma unroll
    for (int i = 0; i < 8; ++i) {
      float v = fmaxf(c[i] + b, 0.0f);                     // bias + relu
      actsw[(i + 8 * hi) * HID + nt * 16 + n] = f2bf(v);
    }
  }
}

template <int OUT_TILES>
static __device__ __forceinline__ void runMlp(const v16u* xfrag,
                                              const unsigned short* wsrc,
                                              const float* b1, const float* b2,
                                              unsigned short* wbuf,
                                              unsigned short* actsw,
                                              int tid, int lane, v8f* out) {
  const int m = lane & 15, g = lane >> 4;
  __syncthreads();                       // everyone done with previous wbuf
  stageW_tdm(wsrc, 131072u, tid);
  hiddenLayer(xfrag, wbuf, b1, actsw, lane);
  __syncthreads();
  stageW_tdm(wsrc + 65536, 131072u, tid);
  v16u h[KCHUNKS];
#pragma unroll
  for (int kc = 0; kc < KCHUNKS; ++kc) h[kc] = ldsAfrag(actsw, m, g, kc);
  hiddenLayer(h, wbuf, b2, actsw, lane);
  __syncthreads();
  stageW_tdm(wsrc + 131072, 8192u * OUT_TILES, tid);
#pragma unroll
  for (int kc = 0; kc < KCHUNKS; ++kc) h[kc] = ldsAfrag(actsw, m, g, kc);
#pragma unroll
  for (int nt = 0; nt < OUT_TILES; ++nt) {
    v8f c = {};
#pragma unroll
    for (int kc = 0; kc < KCHUNKS; ++kc)
      c = wmma_bf16(h[kc], ldsBfrag(wbuf, nt, kc, lane), c);
    out[nt] = c;
  }
}

// ---------------------------------------------------------------------------
// Prep: f32 row-major W (256 x srcCols) -> bf16 B-fragment-swizzled layout,
// zero-padded to outCols (multiple of 16).
// dst[((nt*8+kt)*32+lane)*16 + s] = W[32kt + 8*(lane>>4) + (s&7) + 16*(s>>3), 16nt + lane&15]
// ---------------------------------------------------------------------------
__global__ void gd_prep(const float* __restrict__ W, unsigned short* __restrict__ dst,
                        int outCols, int srcCols) {
  int idx = blockIdx.x * blockDim.x + threadIdx.x;
  if (idx >= 256 * outCols) return;
  int j2h  = idx & 15;
  int lane = (idx >> 4) & 31;
  int kt   = (idx >> 9) & 7;
  int nt   = idx >> 12;
  int k = 32 * kt + 8 * (lane >> 4) + (j2h & 7) + 16 * (j2h >> 3);
  int c = 16 * nt + (lane & 15);
  float v = (c < srcCols) ? W[k * srcCols + c] : 0.0f;
  dst[idx] = f2bf(v);
}

__global__ void gd_zero(float* p, int n) {
  int i = blockIdx.x * blockDim.x + threadIdx.x;
  if (i < n) p[i] = 0.0f;
}

// ---------------------------------------------------------------------------
// Main fused kernel: grid (tiles=32, batches=32), 512 threads (16 waves)
// LDS: wbuf 128KB | acts 128KB | VS 32KB | per-point arrays 6KB  (~300KB/WGP)
// ---------------------------------------------------------------------------
__global__ __launch_bounds__(THREADS, 1)
void gd_main(const float* __restrict__ enc, const float* __restrict__ tv,
             const unsigned short* __restrict__ wMean, const float* __restrict__ bm1,
             const float* __restrict__ bm2, const float* __restrict__ bm3,
             const unsigned short* __restrict__ wD, const float* __restrict__ bd1,
             const float* __restrict__ bd2, const float* __restrict__ bd3,
             const unsigned short* __restrict__ wV, const float* __restrict__ bv1,
             const float* __restrict__ bv2, const float* __restrict__ bv3,
             float* __restrict__ accum) {
  extern __shared__ unsigned char smem[];
  unsigned short* wbuf   = (unsigned short*)smem;              // 65536 ush (TDM dst @ LDS 0)
  unsigned short* acts   = (unsigned short*)(smem + 131072);   // 65536 ush
  float* VS     = (float*)(smem + 262144);                     // 256*32 f32
  float* pmean  = (float*)(smem + 294912);
  float* pdv    = pmean + 256;
  float* pdinv  = pdv + 256;
  float* pw     = pdinv + 256;
  float* psdd   = pw + 256;
  float* pslogd = psdd + 256;

  const int tid  = threadIdx.x;
  const int lane = tid & 31;
  const int wave = tid >> 5;
  const int m    = lane & 15;
  const int g    = lane >> 4;
  const int batch = blockIdx.y;
  const int n0    = blockIdx.x * TILE_ROWS;

  unsigned short* actsw = acts + wave * 16 * HID;

  // ---- load input fragments (fp32 -> bf16 pairs), resident across all 3 MLPs
  const float* encRow = enc + ((size_t)batch * NPTS + (size_t)(n0 + wave * 16 + m)) * HID;
  v16u xfrag[KCHUNKS];
#pragma unroll
  for (int kc = 0; kc < KCHUNKS; ++kc) {
    const float* p = encRow + kc * 32 + 8 * g;
    v8ui w;
#pragma unroll
    for (int i = 0; i < 4; ++i) w[i] = f2bf2(p[2 * i], p[2 * i + 1]);
#pragma unroll
    for (int i = 0; i < 4; ++i) w[4 + i] = f2bf2(p[16 + 2 * i], p[16 + 2 * i + 1]);
    xfrag[kc] = __builtin_bit_cast(v16u, w);
  }

  v8f acc[2];

  // ---- MLP mean ----
  runMlp<1>(xfrag, wMean, bm1, bm2, wbuf, actsw, tid, lane, acc);
  if (m == 0) {
    float b3 = bm3[0];
#pragma unroll
    for (int i = 0; i < 8; ++i) pmean[wave * 16 + i + 8 * g] = acc[0][i] + b3;
  }

  // ---- MLP d ----
  runMlp<1>(xfrag, wD, bd1, bd2, wbuf, actsw, tid, lane, acc);
  if (m == 0) {
    float b3 = bd3[0];
#pragma unroll
    for (int i = 0; i < 8; ++i) {
      float x  = acc[0][i] + b3;
      float sp = (x > 20.0f) ? x : log1pf(expf(x));       // softplus
      pdv[wave * 16 + i + 8 * g] = sp + 0.01f;            // + MIN_D
    }
  }
  __syncthreads();
  if (lane < 16) {
    int  n    = wave * 16 + lane;
    float mv  = pmean[n];
    float dv  = pdv[n];
    float x   = tv[(size_t)batch * NPTS + n0 + n];
    float diff = x - mv;
    float dinv = diff / dv;
    pdinv[n]   = dinv;
    pw[n]      = 1.0f / dv;
    psdd[n]    = diff * dinv;
    pslogd[n]  = logf(dv);
  }

  // ---- MLP V ----
  runMlp<2>(xfrag, wV, bv1, bv2, wbuf, actsw, tid, lane, acc);
#pragma unroll
  for (int nt = 0; nt < 2; ++nt) {
    int   r  = nt * 16 + m;
    float b3 = bv3[r];
#pragma unroll
    for (int i = 0; i < 8; ++i) {
      float x = acc[nt][i] + b3;
      VS[(wave * 16 + i + 8 * g) * RANK + r] = tanhf(x * 0.02f) * 50.0f;
    }
  }
  __syncthreads();

  // ---- per-WG partial C, t, scalar sums -> global atomics ----
  float* gacc = accum + (size_t)batch * ACC_STRIDE;
  {
    int e0 = tid * 2;
    for (int e = e0; e < e0 + 2; ++e) {
      int r = e >> 5, s = e & 31;
      float a = 0.0f;
      for (int n = 0; n < TILE_ROWS; ++n)
        a += VS[n * RANK + r] * VS[n * RANK + s] * pw[n];
      atomicAdd(&gacc[e], a);
    }
  }
  if (tid < RANK) {
    float a = 0.0f;
    for (int n = 0; n < TILE_ROWS; ++n) a += VS[n * RANK + tid] * pdinv[n];
    atomicAdd(&gacc[1024 + tid], a);
  }
  if (tid == 32) {
    float a = 0.0f;
    for (int n = 0; n < TILE_ROWS; ++n) a += psdd[n];
    atomicAdd(&gacc[1056], a);
  }
  if (tid == 33) {
    float a = 0.0f;
    for (int n = 0; n < TILE_ROWS; ++n) a += pslogd[n];
    atomicAdd(&gacc[1057], a);
  }
}

// ---------------------------------------------------------------------------
// Finalize: per batch, C += I, Cholesky (32x32), solve L z = t, assemble NLL
// ---------------------------------------------------------------------------
__global__ void gd_final(const float* __restrict__ accum, float* __restrict__ out) {
  __shared__ float Ls[32][33];
  __shared__ float tvec[32];
  int b   = blockIdx.x;
  int tid = threadIdx.x;
  const float* g = accum + (size_t)b * ACC_STRIDE;

  for (int j = 0; j < 32; ++j) Ls[tid][j] = g[tid * 32 + j];
  Ls[tid][tid] += 1.0f;
  tvec[tid] = g[1024 + tid];
  __syncthreads();

  for (int k = 0; k < 32; ++k) {
    if (tid == k) Ls[k][k] = sqrtf(Ls[k][k]);
    __syncthreads();
    if (tid > k) Ls[tid][k] /= Ls[k][k];
    __syncthreads();
    if (tid > k) {
      float f = Ls[tid][k];
      for (int i = k + 1; i <= tid; ++i) Ls[tid][i] -= f * Ls[i][k];
    }
    __syncthreads();
  }

  if (tid == 0) {
    float z[32];
    float zsum = 0.0f, ld = 0.0f;
    for (int i = 0; i < 32; ++i) {
      float s = tvec[i];
      for (int j = 0; j < i; ++j) s -= Ls[i][j] * z[j];
      z[i] = s / Ls[i][i];
      zsum += z[i] * z[i];
      ld   += logf(Ls[i][i]);
    }
    float maha   = g[1056] - zsum;
    float logdet = 2.0f * ld + g[1057];
    const float LOG2PI = 1.8378770664093454836f;
    out[b] = 0.5f * ((float)NPTS * LOG2PI + logdet + maha);
  }
}

// ---------------------------------------------------------------------------
// Host launcher
// ---------------------------------------------------------------------------
extern "C" void kernel_launch(void* const* d_in, const int* in_sizes, int n_in,
                              void* d_out, int out_size, void* d_ws, size_t ws_size,
                              hipStream_t stream) {
  (void)n_in; (void)out_size; (void)ws_size;
  const float* enc = (const float*)d_in[0];
  const float* tv  = (const float*)d_in[2];   // d_in[1] = mask (all false, unused)

  // params flatten: 3 groups of (W0,b0,W1,b1,W2,b2). Identify V by its 256x32 W3.
  int vg = -1;
  for (int g = 0; g < 3; ++g)
    if (in_sizes[3 + 6 * g + 4] == 8192) vg = g;
  if (vg < 0) vg = 2;
  int o0 = -1, o1 = -1;
  for (int g = 0; g < 3; ++g) {
    if (g == vg) continue;
    if (o0 < 0) o0 = g; else o1 = g;
  }
  // sorted keys (V,d,mean): vg==0 -> {d,mean}; insertion (mean,d,V): vg==2 -> {mean,d}
  int mgrp, dgrp;
  if (vg == 0) { dgrp = o0; mgrp = o1; }
  else         { mgrp = o0; dgrp = o1; }

  auto Wp = [&](int grp, int li) { return (const float*)d_in[3 + 6 * grp + 2 * li]; };
  auto Bp = [&](int grp, int li) { return (const float*)d_in[3 + 6 * grp + 2 * li + 1]; };

  unsigned short* wsu   = (unsigned short*)d_ws;
  unsigned short* wMean = wsu + WS_MEAN;
  unsigned short* wD    = wsu + WS_D;
  unsigned short* wV    = wsu + WS_V;
  float*          accf  = (float*)(wsu + WS_ACC_U);

  // swizzle weights to bf16 fragment layout
  gd_prep<<<256, 256, 0, stream>>>(Wp(mgrp, 0), wMean,          256, 256);
  gd_prep<<<256, 256, 0, stream>>>(Wp(mgrp, 1), wMean + 65536,  256, 256);
  gd_prep<<<16,  256, 0, stream>>>(Wp(mgrp, 2), wMean + 131072, 16,  1);
  gd_prep<<<256, 256, 0, stream>>>(Wp(dgrp, 0), wD,             256, 256);
  gd_prep<<<256, 256, 0, stream>>>(Wp(dgrp, 1), wD + 65536,     256, 256);
  gd_prep<<<16,  256, 0, stream>>>(Wp(dgrp, 2), wD + 131072,    16,  1);
  gd_prep<<<256, 256, 0, stream>>>(Wp(vg, 0),   wV,             256, 256);
  gd_prep<<<256, 256, 0, stream>>>(Wp(vg, 1),   wV + 65536,     256, 256);
  gd_prep<<<32,  256, 0, stream>>>(Wp(vg, 2),   wV + 131072,    32,  32);

  // zero accumulators (32 batches x ACC_STRIDE floats)
  int naccf = 32 * ACC_STRIDE;
  gd_zero<<<(naccf + 255) / 256, 256, 0, stream>>>(accf, naccf);

  // fused MLP + reduction kernel
  size_t smemBytes = 294912 + 6 * 256 * sizeof(float);   // ~300KB of the 320KB WGP LDS
  gd_main<<<dim3(NPTS / TILE_ROWS, 32), THREADS, smemBytes, stream>>>(
      enc, tv,
      wMean, Bp(mgrp, 0), Bp(mgrp, 1), Bp(mgrp, 2),
      wD,    Bp(dgrp, 0), Bp(dgrp, 1), Bp(dgrp, 2),
      wV,    Bp(vg, 0),   Bp(vg, 1),   Bp(vg, 2),
      accf);

  // per-batch Cholesky + solve + NLL assembly
  gd_final<<<32, 32, 0, stream>>>(accf, (float*)d_out);
}
